// MultiHeadedSelfAttention_67637144977651
// MI455X (gfx1250) — matrix operand order repaired and checked by reading, not compiled, not run on hardware
//
#include <hip/hip_runtime.h>

// Problem constants (from the reference)
#define B_  2
#define SQ_ 2048
#define SK_ 4096
#define D_  512
#define H_  8
#define W_  64

typedef __bf16 bf16_t;
typedef __attribute__((ext_vector_type(16))) __bf16 v16bf;
typedef __attribute__((ext_vector_type(8)))  __bf16 v8bf;
typedef __attribute__((ext_vector_type(8)))  float  v8f;

#define WMMA_BF16(a, b, c) \
  __builtin_amdgcn_wmma_f32_16x16x32_bf16(false, (a), false, (b), (short)0, (c), false, false)

__device__ __forceinline__ v8f zero8() {
  v8f z = {0.f, 0.f, 0.f, 0.f, 0.f, 0.f, 0.f, 0.f};
  return z;
}

__device__ __forceinline__ v16bf combine8(v8bf lo, v8bf hi) {
  return __builtin_shufflevector(lo, hi, 0, 1, 2, 3, 4, 5, 6, 7, 8, 9, 10, 11, 12, 13, 14, 15);
}

// A fragment (16x32 bf16), row-major source with leading dim ld.
// Lane L: row m0+(L&15); k runs [k0 + (L>>4)*8 .. +7] and [k0+16+(L>>4)*8 .. +7].
__device__ __forceinline__ v16bf load_a_frag(const bf16_t* __restrict__ src, int ld, int m0, int k0) {
  const int lane = threadIdx.x & 31;
  const int m    = m0 + (lane & 15);
  const int kg   = lane >> 4;
  const bf16_t* p = src + (size_t)m * ld + k0 + kg * 8;
  v8bf lo = *(const v8bf*)p;
  v8bf hi = *(const v8bf*)(p + 16);
  return combine8(lo, hi);
}

// B fragment (32x16): B[k][n] = src[n][k] with k contiguous in memory (leading dim ld).
// Lane L: col n0+(L&15); k runs [k0 + (L>>4)*8 .. +7] and [k0+16+(L>>4)*8 .. +7].
__device__ __forceinline__ v16bf load_b_frag(const bf16_t* __restrict__ src, size_t ld, int n0, int k0) {
  const int lane = threadIdx.x & 31;
  const int n    = n0 + (lane & 15);
  const int kg   = lane >> 4;
  const bf16_t* p = src + (size_t)n * ld + k0 + kg * 8;
  return combine8(*(const v8bf*)p, *(const v8bf*)(p + 16));
}

// ---------------------------------------------------------------------------
// fp32 -> bf16 conversion
__global__ __launch_bounds__(256) void cvt_f32_bf16(const float* __restrict__ src,
                                                    bf16_t* __restrict__ dst, int n) {
  int i = blockIdx.x * 256 + threadIdx.x;
  if (i < n) dst[i] = (bf16_t)src[i];
}

// int mask -> additive float in log2 domain: 0 if keep, -10000*log2(e) if masked
__global__ __launch_bounds__(256) void mask_to_float(const int* __restrict__ m,
                                                     float* __restrict__ mf, int n) {
  int i = blockIdx.x * 256 + threadIdx.x;
  if (i < n) mf[i] = m[i] ? 0.0f : -14426.95f;
}

// 512x512 weight transpose + convert: WT[n][k] = W[k][n]
__global__ __launch_bounds__(256) void transpose_cvt_512(const float* __restrict__ Wsrc,
                                                         bf16_t* __restrict__ WT) {
  int i = blockIdx.x * 256 + threadIdx.x;  // i = k*512 + n
  int k = i >> 9, n = i & 511;
  WT[(size_t)n * 512 + k] = (bf16_t)Wsrc[i];
}

// ---------------------------------------------------------------------------
// Projection GEMM: out[M,512] = A[M,512] @ W[512,512] + bias, bf16 WMMA.
// Each wave: one 16(M) x 64(N) tile; 8 waves/block cover 512 N; grid = M/16.
// MODE 0: row-major bf16 store with scale. MODE 1: store transposed [B][512][SK].
template <int MODE>
__global__ __launch_bounds__(256) void proj_gemm(const bf16_t* __restrict__ A,
                                                 const bf16_t* __restrict__ WT,
                                                 const float* __restrict__ bias,
                                                 bf16_t* __restrict__ out, float scale) {
  const int wid  = threadIdx.x >> 5;   // 0..7 -> N tile of 64
  const int lane = threadIdx.x & 31;
  const int m0   = blockIdx.x * 16;
  const int n0   = wid * 64;

  v8f c[4];
#pragma unroll
  for (int nb = 0; nb < 4; ++nb) c[nb] = zero8();

  for (int kb = 0; kb < D_; kb += 32) {
    v16bf a = load_a_frag(A, D_, m0, kb);
#pragma unroll
    for (int nb = 0; nb < 4; ++nb) {
      v16bf b = load_b_frag(WT, D_, n0 + nb * 16, kb);
      c[nb] = WMMA_BF16(a, b, c[nb]);
    }
  }

  const int nlo  = lane & 15;
  const int half = lane >> 4;
#pragma unroll
  for (int nb = 0; nb < 4; ++nb) {
    const int d   = n0 + nb * 16 + nlo;
    const float bv = bias[d];
    if (MODE == 0) {
#pragma unroll
      for (int r = 0; r < 8; ++r) {
        const int row = m0 + r + half * 8;
        out[(size_t)row * D_ + d] = (bf16_t)((c[nb][r] + bv) * scale);
      }
    } else {
      // vpT layout [B][512][SK]; per-lane 8 rows are contiguous sk -> one 16B store
      const int b   = m0 >> 12;            // m0 / SK_
      const int skl = (m0 & (SK_ - 1)) + half * 8;
      v8bf pk;
#pragma unroll
      for (int r = 0; r < 8; ++r) pk[r] = (bf16_t)((c[nb][r] + bv) * scale);
      *(v8bf*)(out + ((size_t)b * D_ + d) * SK_ + skl) = pk;
    }
  }
}

// ---------------------------------------------------------------------------
// Attention: wave = 16 q-rows of one (b,h); SK streamed in 64-col chunks.
// qp is pre-scaled by (1/sqrt(64))*log2(e) so p = exp2(score + maskadd), using
// raw v_exp_f32 (hardware exp2; deep-negative masked inputs flush to exact 0).
// Fixed-reference softmax (scores bounded, no overflow possible), so no
// running max / rescale. Row sums are accumulated with WMMA against an
// all-ones B matrix: every lane of lsum row r holds the row sum (broadcast
// layout), ready for the final divide.
__global__ __launch_bounds__(256) void attn_kernel(const bf16_t* __restrict__ qp,
                                                   const bf16_t* __restrict__ kp,
                                                   const bf16_t* __restrict__ vpT,
                                                   const float* __restrict__ maskf,
                                                   float* __restrict__ out) {
  __shared__ __align__(32) bf16_t plds[8][16 * 64];  // per-wave private P staging

  const int wid  = threadIdx.x >> 5;
  const int lane = threadIdx.x & 31;
  const int wg   = blockIdx.x * 8 + wid;     // global wave-tile id
  const int mt   = wg & 127;                 // 128 q-tiles per (b,h)
  const int bh   = wg >> 7;                  // 0..15
  const int b    = bh >> 3;
  const int hh   = bh & 7;
  const int q0   = mt * 16;
  const int nlo  = lane & 15;
  const int kg   = lane >> 4;
  const int half = kg;

  const bf16_t* qbase = qp + (size_t)b * SQ_ * D_ + hh * W_;
  const bf16_t* kbase = kp + (size_t)b * SK_ * D_ + hh * W_;
  const bf16_t* vbase = vpT + ((size_t)b * D_ + hh * W_) * SK_;
  const float*  mbase = maskf + b * SK_;

  // Q fragments (W=64 -> two 16x32 A frags), loaded once
  v16bf a0 = load_a_frag(qbase, D_, q0, 0);
  v16bf a1 = load_a_frag(qbase, D_, q0, 32);

  // all-ones B fragment for WMMA row sums
  v16bf ones;
#pragma unroll
  for (int e = 0; e < 16; ++e) ones[e] = (bf16_t)1.0f;

  v8f lsum = zero8();
  v8f o[4];
#pragma unroll
  for (int nb = 0; nb < 4; ++nb) o[nb] = zero8();

  bf16_t* pl = &plds[wid][0];

  for (int sk0 = 0; sk0 < SK_; sk0 += 64) {
    // ---- scores (16x64) + mask + exp2, straight into LDS as bf16 P ----
#pragma unroll
    for (int nb = 0; nb < 4; ++nb) {
      v16bf b0 = load_b_frag(kbase, D_, sk0 + nb * 16, 0);
      v16bf b1 = load_b_frag(kbase, D_, sk0 + nb * 16, 32);
      v8f acc = zero8();
      acc = WMMA_BF16(a0, b0, acc);
      acc = WMMA_BF16(a1, b1, acc);
      const float madd = mbase[sk0 + nb * 16 + nlo];
#pragma unroll
      for (int r = 0; r < 8; ++r) {
        const float p = __builtin_amdgcn_exp2f(acc[r] + madd);  // raw v_exp_f32
        pl[(r + half * 8) * 64 + nb * 16 + nlo] = (bf16_t)p;
      }
    }

    // ---- reload P as A fragments (per-wave private region; LDS is in-order
    //      per wave, so no cross-wave barrier is required) ----
    v8bf plo0 = *(const v8bf*)(pl + nlo * 64 + 0 + kg * 8);
    v8bf phi0 = *(const v8bf*)(pl + nlo * 64 + 16 + kg * 8);
    v8bf plo1 = *(const v8bf*)(pl + nlo * 64 + 32 + kg * 8);
    v8bf phi1 = *(const v8bf*)(pl + nlo * 64 + 48 + kg * 8);
    v16bf pa0 = combine8(plo0, phi0);
    v16bf pa1 = combine8(plo1, phi1);

    // ---- row sums via WMMA against ones ----
    lsum = WMMA_BF16(pa0, ones, lsum);
    lsum = WMMA_BF16(pa1, ones, lsum);

    // ---- PV: o[16x64] += P[16x64] @ V[64x64] ----
#pragma unroll
    for (int nb = 0; nb < 4; ++nb) {
      v16bf bv0 = load_b_frag(vbase, SK_, nb * 16, sk0);
      v16bf bv1 = load_b_frag(vbase, SK_, nb * 16, sk0 + 32);
      o[nb] = WMMA_BF16(pa0, bv0, o[nb]);
      o[nb] = WMMA_BF16(pa1, bv1, o[nb]);
    }
  }

  // ---- epilogue: divide by row sums (broadcast layout), store fp32 ----
  float* obase = out + ((size_t)b * SQ_ + q0) * D_ + hh * W_;
  float inv[8];
#pragma unroll
  for (int r = 0; r < 8; ++r) inv[r] = __builtin_amdgcn_rcpf(lsum[r]);  // v_rcp_f32
#pragma unroll
  for (int nb = 0; nb < 4; ++nb)
#pragma unroll
    for (int r = 0; r < 8; ++r)
      obase[(size_t)(r + half * 8) * D_ + nb * 16 + nlo] = o[nb][r] * inv[r];
}

// ---------------------------------------------------------------------------
extern "C" void kernel_launch(void* const* d_in, const int* in_sizes, int n_in,
                              void* d_out, int out_size, void* d_ws, size_t ws_size,
                              hipStream_t stream) {
  const float* q   = (const float*)d_in[0];
  const float* k   = (const float*)d_in[1];
  const int*   msk = (const int*)d_in[2];
  const float* Wq  = (const float*)d_in[3];
  const float* bq  = (const float*)d_in[4];
  const float* Wk  = (const float*)d_in[5];
  const float* bk  = (const float*)d_in[6];
  const float* Wv  = (const float*)d_in[7];
  const float* bv  = (const float*)d_in[8];
  float* out = (float*)d_out;

  char* ws = (char*)d_ws;
  size_t off = 0;
  auto carveB = [&](size_t bytes) {
    void* p = ws + off;
    off += ((bytes + 255) & ~(size_t)255);
    return p;
  };
  bf16_t* qbf  = (bf16_t*)carveB((size_t)B_ * SQ_ * D_ * 2);  // q in bf16
  bf16_t* kbf  = (bf16_t*)carveB((size_t)B_ * SK_ * D_ * 2);  // k in bf16
  bf16_t* WqT  = (bf16_t*)carveB((size_t)D_ * D_ * 2);        // transposed weights
  bf16_t* WkT  = (bf16_t*)carveB((size_t)D_ * D_ * 2);
  bf16_t* WvT  = (bf16_t*)carveB((size_t)D_ * D_ * 2);
  bf16_t* qpw  = (bf16_t*)carveB((size_t)B_ * SQ_ * D_ * 2);  // q-proj (pre-scaled)
  bf16_t* kpw  = (bf16_t*)carveB((size_t)B_ * SK_ * D_ * 2);  // k-proj, row-major
  bf16_t* vpT  = (bf16_t*)carveB((size_t)B_ * D_ * SK_ * 2);  // v-proj, [B][D][SK]
  float*  mskf = (float*)carveB((size_t)B_ * SK_ * 4);        // additive mask

  const int nq = B_ * SQ_ * D_;
  const int nk = B_ * SK_ * D_;
  const int nm = B_ * SK_;
  cvt_f32_bf16<<<(nq + 255) / 256, 256, 0, stream>>>(q, qbf, nq);
  cvt_f32_bf16<<<(nk + 255) / 256, 256, 0, stream>>>(k, kbf, nk);
  mask_to_float<<<(nm + 255) / 256, 256, 0, stream>>>(msk, mskf, nm);
  transpose_cvt_512<<<(D_ * D_) / 256, 256, 0, stream>>>(Wq, WqT);
  transpose_cvt_512<<<(D_ * D_) / 256, 256, 0, stream>>>(Wk, WkT);
  transpose_cvt_512<<<(D_ * D_) / 256, 256, 0, stream>>>(Wv, WvT);

  // Fold 1/sqrt(64) and log2(e) into the q projection.
  const float qscale = 0.125f * 1.44269504f;
  proj_gemm<0><<<(B_ * SQ_) / 16, 256, 0, stream>>>(qbf, WqT, bq, qpw, qscale);
  proj_gemm<0><<<(B_ * SK_) / 16, 256, 0, stream>>>(kbf, WkT, bk, kpw, 1.0f);
  proj_gemm<1><<<(B_ * SK_) / 16, 256, 0, stream>>>(kbf, WvT, bv, vpT, 1.0f);

  // 2048 wave-tiles (16 q-rows each) / 8 waves per block = 256 blocks.
  attn_kernel<<<(B_ * H_ * (SQ_ / 16)) / 8, 256, 0, stream>>>(qpw, kpw, vpT, mskf, out);
}
